// GraphSAGE_11493332484323
// MI455X (gfx1250) — compile-verified, matrix operand order checked by reading, not run on hardware
//
#include <hip/hip_runtime.h>
#include <hip/hip_bf16.h>

typedef __attribute__((ext_vector_type(16))) __bf16 v16bf;
typedef __attribute__((ext_vector_type(8)))  float  v8f;
typedef __attribute__((ext_vector_type(4)))  unsigned int u32x4;

union BFrag {
    v16bf v;
    unsigned short u[16];
    u32x4 q[2];
};

__device__ __forceinline__ unsigned short f2bf_rne(float f) {
    unsigned int u = __float_as_uint(f);
    u += 0x7FFFu + ((u >> 16) & 1u);
    return (unsigned short)(u >> 16);
}
__device__ __forceinline__ float bf2f(unsigned short h) {
    return __uint_as_float((unsigned int)h << 16);
}

// ---------------------------------------------------------------------------
// Kernel 1: split W = [W1_l | W1_r] (row-major [k][n]) into column-major
// bf16 hi/lo planes: Wh[n][k], Wl[n][k].  256 cols x 128 k.
// ---------------------------------------------------------------------------
__global__ void prep_w_kernel(const float* __restrict__ W1l,
                              const float* __restrict__ W1r,
                              unsigned short* __restrict__ Wh,
                              unsigned short* __restrict__ Wl) {
    int t = blockIdx.x * blockDim.x + threadIdx.x;     // 0..32767
    int n = t >> 7;                                    // 0..255 (output column)
    int k = t & 127;                                   // 0..127
    float f = (n < 128) ? W1l[k * 128 + n] : W1r[k * 128 + (n - 128)];
    unsigned short hi = f2bf_rne(f);
    Wh[n * 128 + k] = hi;
    Wl[n * 128 + k] = f2bf_rne(f - bf2f(hi));
}

// ---------------------------------------------------------------------------
// Kernel 2: Y[N,256] = X[N,128] @ W[128,256] via bf16x3 WMMA, f32 accumulate.
// One wave per 16-row tile; loops 16 column tiles; K=128 -> 4 ksteps x 3 WMMA.
// A layout (16-bit, 16x32): lanes 0-15 row M=lane, K 0..7 & 16..23;
//                           lanes 16-31 row M=lane-16, K 8..15 & 24..31.
// B layout mirrored: lane = column, same K striping.
// C/D: VGPR i -> row (i + 8*half), col = lane%16.
// Tail handling is a single wave-uniform branch so the hot path has no
// per-store exec-mask manipulation.
// ---------------------------------------------------------------------------
__global__ void gemm_y_kernel(const float* __restrict__ x,
                              const unsigned short* __restrict__ Wh,
                              const unsigned short* __restrict__ Wl,
                              float* __restrict__ Y,
                              int nTiles, int N) {
    int wid = blockIdx.x * (blockDim.x >> 5) + (threadIdx.x >> 5);
    if (wid >= nTiles) return;                         // wave-uniform: EXEC stays all-1
    int lane = threadIdx.x & 31;
    int half = lane >> 4;                              // 0 or 1
    int m    = lane & 15;
    int rb   = wid * 16;
    bool fullTile = (rb + 16 <= N);                    // wave-uniform

    int arow = rb + m;
    if (arow >= N) arow = N - 1;                       // tail clamp (loads only)
    const float* xrow = x + (size_t)arow * 128;

    // Load + split A fragments once, reuse across all 16 column tiles.
    BFrag Ah[4], Al[4];
#pragma unroll
    for (int kk = 0; kk < 4; ++kk) {
        int k1 = kk * 32 + half * 8;
        int k2 = k1 + 16;
#pragma unroll
        for (int j = 0; j < 8; ++j) {
            float f = xrow[k1 + j];
            unsigned short hi = f2bf_rne(f);
            Ah[kk].u[j]     = hi;
            Al[kk].u[j]     = f2bf_rne(f - bf2f(hi));
            float g = xrow[k2 + j];
            unsigned short hj = f2bf_rne(g);
            Ah[kk].u[8 + j] = hj;
            Al[kk].u[8 + j] = f2bf_rne(g - bf2f(hj));
        }
    }

#pragma unroll 1
    for (int ct = 0; ct < 16; ++ct) {
        int col = ct * 16 + m;
        const unsigned short* whc = Wh + (size_t)col * 128;
        const unsigned short* wlc = Wl + (size_t)col * 128;

        v8f acc = {};
#pragma unroll
        for (int kk = 0; kk < 4; ++kk) {
            int k1 = kk * 32 + half * 8;               // bytes: *2 -> 16B aligned
            int k2 = k1 + 16;
            BFrag Bh, Bl;
            Bh.q[0] = *(const u32x4*)(whc + k1);
            Bh.q[1] = *(const u32x4*)(whc + k2);
            Bl.q[0] = *(const u32x4*)(wlc + k1);
            Bl.q[1] = *(const u32x4*)(wlc + k2);
            acc = __builtin_amdgcn_wmma_f32_16x16x32_bf16(
                      false, Ah[kk].v, false, Bh.v, (short)0, acc, false, false);
            acc = __builtin_amdgcn_wmma_f32_16x16x32_bf16(
                      false, Ah[kk].v, false, Bl.v, (short)0, acc, false, false);
            acc = __builtin_amdgcn_wmma_f32_16x16x32_bf16(
                      false, Al[kk].v, false, Bh.v, (short)0, acc, false, false);
        }

        float* ybase = Y + (size_t)(rb + half * 8) * 256 + col;
        if (fullTile) {
            // hot path: no per-store predication
#pragma unroll
            for (int i = 0; i < 8; ++i) ybase[(size_t)i * 256] = acc[i];
        } else {
#pragma unroll
            for (int i = 0; i < 8; ++i) {
                if (rb + half * 8 + i < N) ybase[(size_t)i * 256] = acc[i];
            }
        }
    }
}

// ---------------------------------------------------------------------------
// Kernel 3: layer-1 scatter. One wave per edge: agg[dst] += Y_l[src] (128 f32),
// deg[dst] += 1.  Coalesced: lane handles idx = j*32+lane.
// ---------------------------------------------------------------------------
__global__ void scatter1_kernel(const float* __restrict__ Y,
                                const int* __restrict__ src,
                                const int* __restrict__ dst,
                                float* __restrict__ agg,
                                float* __restrict__ deg, int E) {
    int e = blockIdx.x * (blockDim.x >> 5) + (threadIdx.x >> 5);
    if (e >= E) return;
    int lane = threadIdx.x & 31;
    int s = src[e], d = dst[e];
    const float* yrow = Y + (size_t)s * 256;           // Y_l = cols 0..127
    float* arow = agg + (size_t)d * 128;
#pragma unroll
    for (int j = 0; j < 4; ++j) {
        int idx = j * 32 + lane;
        atomicAdd(&arow[idx], yrow[idx]);
    }
    if (lane == 0) atomicAdd(&deg[d], 1.0f);
}

// ---------------------------------------------------------------------------
// Kernel 4: per node, h = relu(agg/max(deg,1) + Y_r + b1) (never stored);
// emit s = h.W2_l and t = h.W2_r.  One wave per node, butterfly reduction.
// ---------------------------------------------------------------------------
__global__ void combine_kernel(const float* __restrict__ agg,
                               const float* __restrict__ Y,
                               const float* __restrict__ deg,
                               const float* __restrict__ b1,
                               const float* __restrict__ W2l,
                               const float* __restrict__ W2r,
                               float* __restrict__ sarr,
                               float* __restrict__ tarr, int N) {
    int i = blockIdx.x * (blockDim.x >> 5) + (threadIdx.x >> 5);
    if (i >= N) return;
    int lane = threadIdx.x & 31;
    float inv = 1.0f / fmaxf(deg[i], 1.0f);
    const float* arow = agg + (size_t)i * 128;
    const float* yr   = Y + (size_t)i * 256 + 128;     // Y_r = cols 128..255
    float sv = 0.0f, tv = 0.0f;
#pragma unroll
    for (int j = 0; j < 4; ++j) {
        int idx = j * 32 + lane;
        float h = fmaxf(arow[idx] * inv + yr[idx] + b1[idx], 0.0f);
        sv += h * W2l[idx];
        tv += h * W2r[idx];
    }
#pragma unroll
    for (int off = 16; off >= 1; off >>= 1) {
        sv += __shfl_xor(sv, off, 32);
        tv += __shfl_xor(tv, off, 32);
    }
    if (lane == 0) { sarr[i] = sv; tarr[i] = tv; }
}

// ---------------------------------------------------------------------------
// Kernel 5: layer-2 scatter, ONE scalar atomic per edge (128x traffic cut).
// ---------------------------------------------------------------------------
__global__ void scatter2_kernel(const int* __restrict__ src,
                                const int* __restrict__ dst,
                                const float* __restrict__ sarr,
                                float* __restrict__ sacc, int E) {
    int e = blockIdx.x * blockDim.x + threadIdx.x;
    if (e >= E) return;
    atomicAdd(&sacc[dst[e]], sarr[src[e]]);
}

// ---------------------------------------------------------------------------
// Kernel 6: out[i] = sacc/max(deg,1) + t + b2
// ---------------------------------------------------------------------------
__global__ void final_kernel(const float* __restrict__ sacc,
                             const float* __restrict__ deg,
                             const float* __restrict__ tarr,
                             const float* __restrict__ b2,
                             float* __restrict__ out, int N) {
    int i = blockIdx.x * blockDim.x + threadIdx.x;
    if (i >= N) return;
    out[i] = sacc[i] / fmaxf(deg[i], 1.0f) + tarr[i] + b2[0];
}

extern "C" void kernel_launch(void* const* d_in, const int* in_sizes, int n_in,
                              void* d_out, int out_size, void* d_ws, size_t ws_size,
                              hipStream_t stream) {
    const float* x   = (const float*)d_in[0];
    const int*   ei  = (const int*)d_in[1];
    const float* W1l = (const float*)d_in[2];
    const float* b1  = (const float*)d_in[3];
    const float* W1r = (const float*)d_in[4];
    const float* W2l = (const float*)d_in[5];
    const float* b2  = (const float*)d_in[6];
    const float* W2r = (const float*)d_in[7];
    float* out = (float*)d_out;

    const int N = in_sizes[0] / 128;
    const int E = in_sizes[1] / 2;
    const int* src = ei;
    const int* dst = ei + E;

    // workspace carve-up (256B-aligned regions)
    char* ws = (char*)d_ws;
    size_t off = 0;
    auto carve = [&](size_t bytes) {
        void* p = ws + off;
        off += (bytes + 255) & ~(size_t)255;
        return p;
    };
    unsigned short* Wh = (unsigned short*)carve((size_t)256 * 128 * 2);
    unsigned short* Wl = (unsigned short*)carve((size_t)256 * 128 * 2);
    float* Y    = (float*)carve((size_t)N * 256 * 4);
    float* agg  = (float*)carve((size_t)N * 128 * 4);
    float* deg  = (float*)carve((size_t)N * 4);
    float* sarr = (float*)carve((size_t)N * 4);
    float* tarr = (float*)carve((size_t)N * 4);
    float* sacc = (float*)carve((size_t)N * 4);

    // accumulators must be zeroed every call (graph-capture safe)
    hipMemsetAsync(agg,  0, (size_t)N * 128 * 4, stream);
    hipMemsetAsync(deg,  0, (size_t)N * 4, stream);
    hipMemsetAsync(sacc, 0, (size_t)N * 4, stream);

    prep_w_kernel<<<128, 256, 0, stream>>>(W1l, W1r, Wh, Wl);

    int nTiles = (N + 15) / 16;                        // 3125
    gemm_y_kernel<<<(nTiles + 7) / 8, 256, 0, stream>>>(x, Wh, Wl, Y, nTiles, N);

    scatter1_kernel<<<(E + 7) / 8, 256, 0, stream>>>(Y, src, dst, agg, deg, E);

    combine_kernel<<<(N + 7) / 8, 256, 0, stream>>>(agg, Y, deg, b1, W2l, W2r,
                                                    sarr, tarr, N);

    scatter2_kernel<<<(E + 255) / 256, 256, 0, stream>>>(src, dst, sarr, sacc, E);

    final_kernel<<<(N + 255) / 256, 256, 0, stream>>>(sacc, deg, tarr, b2, out, N);
}
